// DeepseekV3Attention_41111426957770
// MI455X (gfx1250) — matrix operand order repaired and checked by reading, not compiled, not run on hardware
//
#include <hip/hip_runtime.h>
#include <hip/hip_bf16.h>

// ---------------------------------------------------------------------------
// Trivial ext-vector types (unions over them stay trivially constructible)
// ---------------------------------------------------------------------------
typedef __attribute__((ext_vector_type(16))) __bf16        v16bf;
typedef __attribute__((ext_vector_type(8)))  float         v8f;
typedef __attribute__((ext_vector_type(4))) unsigned int   u32x4;
typedef __attribute__((ext_vector_type(4)))  float         f32x4;

union Frag { v16bf v; u32x4 u[2]; };

__device__ __forceinline__ unsigned short f32_to_bf16(float f) {
  unsigned int u = __float_as_uint(f);
  unsigned int r = u + 0x7FFFu + ((u >> 16) & 1u);   // RNE
  return (unsigned short)(r >> 16);
}
__device__ __forceinline__ float bf16_to_f32(unsigned short u) {
  return __uint_as_float(((unsigned int)u) << 16);
}
// Packed f32x2 -> bf16x2 (CDNA5 v_cvt_pk_bf16_f32): lo = bf16(a), hi = bf16(b)
__device__ __forceinline__ unsigned int cvt_pk_bf16(float a, float b) {
  unsigned int r;
  asm volatile("v_cvt_pk_bf16_f32 %0, %1, %2" : "=v"(r) : "v"(a), "v"(b));
  return r;
}
// Async global->LDS 16-byte copy (per-lane addresses both sides, ASYNCcnt)
__device__ __forceinline__ void async_copy16(void* lds_ptr, const void* gptr) {
  unsigned int loff = (unsigned int)(size_t)lds_ptr;   // low 32 bits = LDS offset
  asm volatile("global_load_async_to_lds_b128 %0, %1, off"
               :: "v"(loff), "v"(gptr) : "memory");
}
__device__ __forceinline__ void wait_async() {
  asm volatile("s_wait_asynccnt 0" ::: "memory");
}

// ---------------------------------------------------------------------------
// GEMM: C[M,N] = A[M,K] @ B[K,N](f32 weights), bf16 WMMA, f32 accum.
// AT = float (v_cvt_pk_bf16_f32 convert) or ushort (bf16: async-to-LDS).
// Double-buffered LDS, BK=64 (two WMMA k-steps per staged tile).
// 256 thr / 8 waves, tile 128x128. Wave w: 32x64 (2x4 16x16 accum tiles).
// sA[buf] = [128][64] row-major; sB[buf] = [128][64] transposed ([n][k]).
// ---------------------------------------------------------------------------
template <typename AT, bool C_BF16>
__global__ __launch_bounds__(256)
void gemm_wmma(const AT* __restrict__ A, const float* __restrict__ Bm,
               void* __restrict__ Cv, int N, int K, int lda, int ldb, int ldc)
{
  __shared__ __align__(16) unsigned short sA[2][128 * 64];
  __shared__ __align__(16) unsigned short sB[2][128 * 64];   // [n][k]

  const int tid    = threadIdx.x;
  const int w      = tid >> 5;
  const int lane   = tid & 31;
  const int lane15 = lane & 15;
  const int hi     = lane >> 4;
  const int bm     = blockIdx.y * 128;
  const int bn     = blockIdx.x * 128;
  const int waveM  = (w >> 1) * 32;
  const int waveN  = (w & 1) * 64;
  const int lo     = hi * 8;
  const int kb16   = hi * 16;
  constexpr bool A_IS_F32 = (sizeof(AT) == 4);

  v8f acc[2][4];
#pragma unroll
  for (int i = 0; i < 2; ++i)
#pragma unroll
    for (int j = 0; j < 4; ++j) acc[i][j] = v8f{};

  // ---- staging: A tile (128 x 64) + B tile (64 x 128, transposed) ----
  auto stage = [&](int kb, int buf) {
    if (A_IS_F32) {
      const int r = tid >> 1, c0 = (tid & 1) * 32;
      const float* src = (const float*)A + (size_t)(bm + r) * lda + kb + c0;
      unsigned int* dst = (unsigned int*)&sA[buf][r * 64 + c0];
#pragma unroll
      for (int q4 = 0; q4 < 8; ++q4) {
        f32x4 f = *(const f32x4*)(src + q4 * 4);
        dst[q4 * 2 + 0] = cvt_pk_bf16(f[0], f[1]);
        dst[q4 * 2 + 1] = cvt_pk_bf16(f[2], f[3]);
      }
    } else {
      // bf16 A: pure copy, async global->LDS, 4 x 16B per thread
#pragma unroll
      for (int i = 0; i < 4; ++i) {
        const int c = tid * 4 + i;            // 0..1023 chunks of 16B
        const int r = c >> 3, o16 = (c & 7) * 16;
        const char* g = (const char*)A + ((size_t)(bm + r) * lda + kb) * 2 + o16;
        async_copy16((char*)&sA[buf][0] + r * 128 + o16, g);
      }
    }
    // B: column-per-thread so the k-axis pairs pack (cvt_pk + b32 store)
    {
      const int n = tid & 127, k0 = (tid >> 7) * 32;
      const int col = bn + n;
      const bool ok = (col < N);
      const float* src = Bm + (size_t)(kb + k0) * ldb + col;
      unsigned int* dst = (unsigned int*)&sB[buf][n * 64 + k0];
      if (kb + 64 < K) __builtin_prefetch(src + (size_t)64 * ldb, 0, 1);
#pragma unroll
      for (int jj = 0; jj < 16; ++jj) {
        const float a = ok ? src[(size_t)(2 * jj) * ldb] : 0.0f;
        const float b = ok ? src[(size_t)(2 * jj + 1) * ldb] : 0.0f;
        dst[jj] = cvt_pk_bf16(a, b);
      }
    }
  };

  auto compute = [&](int buf) {
#pragma unroll
    for (int ks = 0; ks < 64; ks += 32) {
      Frag af[2], bfr[4];
#pragma unroll
      for (int mt = 0; mt < 2; ++mt) {
        const unsigned short* p = &sA[buf][(waveM + mt * 16 + lane15) * 64 + ks + lo];
        af[mt].u[0] = *(const u32x4*)(p);
        af[mt].u[1] = *(const u32x4*)(p + 16);
      }
#pragma unroll
      for (int nt = 0; nt < 4; ++nt) {
        const unsigned short* p = &sB[buf][(waveN + nt * 16 + lane15) * 64 + ks + lo];
        bfr[nt].u[0] = *(const u32x4*)(p);
        bfr[nt].u[1] = *(const u32x4*)(p + 16);
      }
#pragma unroll
      for (int mt = 0; mt < 2; ++mt)
#pragma unroll
        for (int nt = 0; nt < 4; ++nt)
          acc[mt][nt] = __builtin_amdgcn_wmma_f32_16x16x32_bf16(
              false, af[mt].v, false, bfr[nt].v, (short)0, acc[mt][nt], false, false);
    }
  };

  // ---- double-buffered main loop ----
  const int nk = K >> 6;
  int buf = 0;
  stage(0, 0);
  if (!A_IS_F32) wait_async();
  __syncthreads();
  for (int t = 0; t < nk; ++t) {
    if (t + 1 < nk) stage((t + 1) << 6, buf ^ 1);
    compute(buf);
    if (!A_IS_F32) wait_async();
    __syncthreads();
    buf ^= 1;
  }

  // ---- store C ----
#pragma unroll
  for (int mt = 0; mt < 2; ++mt)
#pragma unroll
    for (int nt = 0; nt < 4; ++nt) {
      const int col = bn + waveN + nt * 16 + lane15;
      if (col < N) {
#pragma unroll
        for (int j = 0; j < 8; ++j) {
          const int row = bm + waveM + mt * 16 + j + 8 * hi;
          if (C_BF16)
            ((unsigned short*)Cv)[(size_t)row * ldc + col] = f32_to_bf16(acc[mt][nt][j]);
          else
            ((float*)Cv)[(size_t)row * ldc + col] = acc[mt][nt][j];
        }
      }
    }
}

// ---------------------------------------------------------------------------
// RMSNorm over bf16 rows (f32 math): y = x * rsqrt(mean(x^2)+eps) * g
// ---------------------------------------------------------------------------
__global__ __launch_bounds__(256)
void rmsnorm_bf16(const unsigned short* __restrict__ x, const float* __restrict__ g,
                  unsigned short* __restrict__ y, int D, int sx, int sy)
{
  const int row = blockIdx.x;
  const unsigned short* xr = x + (size_t)row * sx;
  float ss = 0.f;
  for (int i = threadIdx.x; i < D; i += blockDim.x) {
    float v = bf16_to_f32(xr[i]); ss += v * v;
  }
  __shared__ float red[8];
#pragma unroll
  for (int off = 16; off; off >>= 1) ss += __shfl_down(ss, off, 32);
  const int wid = threadIdx.x >> 5, lane = threadIdx.x & 31;
  if (lane == 0) red[wid] = ss;
  __syncthreads();
  if (wid == 0) {
    float v = (lane < 8) ? red[lane] : 0.f;
#pragma unroll
    for (int off = 4; off; off >>= 1) v += __shfl_down(v, off, 32);
    if (lane == 0) red[0] = v;
  }
  __syncthreads();
  const float inv = rsqrtf(red[0] / (float)D + 1e-6f);
  unsigned short* yr = y + (size_t)row * sy;
  for (int i = threadIdx.x; i < D; i += blockDim.x)
    yr[i] = f32_to_bf16(bf16_to_f32(xr[i]) * inv * g[i]);
}

// ---------------------------------------------------------------------------
// RoPE (DR=64, 32/32 split). inv_freq[d] = 10000^(-d/32)
// ---------------------------------------------------------------------------
__global__ void rope_k_kernel(const unsigned short* __restrict__ kv,
                              unsigned short* __restrict__ kr, int S)
{
  const int row = blockIdx.x, d = threadIdx.x;      // d: 0..31
  const int pos = row % S;
  const float freq = __expf(-(float)d * 0.28782313662425536f);  // ln(1e4)/32
  float sn, cs; __sincosf((float)pos * freq, &sn, &cs);
  const unsigned short* p = kv + (size_t)row * 576 + 512;
  const float x1 = bf16_to_f32(p[d]), x2 = bf16_to_f32(p[d + 32]);
  kr[(size_t)row * 64 + d]      = f32_to_bf16(x1 * cs - x2 * sn);
  kr[(size_t)row * 64 + 32 + d] = f32_to_bf16(x1 * sn + x2 * cs);
}

__global__ void rope_q_kernel(unsigned short* __restrict__ q, int S)
{
  const int row = blockIdx.x;
  const int h = threadIdx.x >> 5, d = threadIdx.x & 31;
  const int pos = row % S;
  const float freq = __expf(-(float)d * 0.28782313662425536f);
  float sn, cs; __sincosf((float)pos * freq, &sn, &cs);
  unsigned short* p = q + (size_t)row * 3072 + 2048 + h * 64;
  const float x1 = bf16_to_f32(p[d]), x2 = bf16_to_f32(p[d + 32]);
  p[d]      = f32_to_bf16(x1 * cs - x2 * sn);
  p[d + 32] = f32_to_bf16(x1 * sn + x2 * cs);
}

// ---------------------------------------------------------------------------
// Flash-style causal MLA attention (all operands bf16 in global memory).
// Block: 128 thr (4 waves); 64-query tile per (b,h,qb); key blocks of 64.
// Q/K staged by async global->LDS gather copies; V staged transposed.
// Softmax row-sums computed on the matrix pipe: rowsum(P) = P @ ones via
// two extra WMMAs on the already-reloaded A-layout P fragments (saves 32
// ds_bpermute per iteration); row-max keeps a 16-lane xor butterfly.
// LDS: sQ 24KB (reused as per-wave P) + sK 24KB + sV 16KB = 64KB.
// ---------------------------------------------------------------------------
__global__ __launch_bounds__(128)
void mla_attn_kernel(const unsigned short* __restrict__ q,     // [M,3072]
                     const unsigned short* __restrict__ kvb,   // [M,4096]
                     const unsigned short* __restrict__ krope, // [M,64]
                     unsigned short* __restrict__ out,         // [M,2048]
                     int S)
{
  const int qb   = blockIdx.x;
  const int h    = blockIdx.y;
  const int b    = blockIdx.z;
  const int tid  = threadIdx.x;
  const int w    = tid >> 5;
  const int lane = tid & 31;
  const int lane15 = lane & 15;
  const int hi   = lane >> 4;
  const int lo   = hi * 8;
  const int kb16 = hi * 16;
  const size_t base = (size_t)b * S;

  __shared__ __align__(16) unsigned short sQ[64 * 192];   // reused as P
  __shared__ __align__(16) unsigned short sK[64 * 192];
  __shared__ __align__(16) unsigned short sV[128 * 64];   // [dv][key]

  // ---- stage Q (64 x 192) via async 16B gather copies (nope ++ rope) ----
  for (int c = tid; c < 64 * 24; c += 128) {              // 24 x 16B per row
    const int r = c / 24, d0 = (c % 24) * 8;
    const size_t tok = base + qb * 64 + r;
    const unsigned short* g = (d0 < 128)
        ? q + tok * 3072 + h * 128 + d0
        : q + tok * 3072 + 2048 + h * 64 + (d0 - 128);
    async_copy16(&sQ[r * 192 + d0], g);
  }
  wait_async();
  __syncthreads();

  // ---- Q fragments (A-layout) ----
  Frag qf[6];
  {
    const int r0 = w * 16 + lane15;
#pragma unroll
    for (int ck = 0; ck < 6; ++ck) {
      const unsigned short* p = &sQ[r0 * 192 + ck * 32 + lo];
      qf[ck].u[0] = *(const u32x4*)(p);
      qf[ck].u[1] = *(const u32x4*)(p + 16);
    }
  }
  __syncthreads();   // sQ now reusable as P

  // all-ones bf16 B fragment for WMMA row sums
  Frag ones;
  {
    u32x4 t = {0x3F803F80u, 0x3F803F80u, 0x3F803F80u, 0x3F803F80u};
    ones.u[0] = t; ones.u[1] = t;
  }

  v8f   o[8];
  float m[8], l[8];
#pragma unroll
  for (int i = 0; i < 8; ++i) { o[i] = v8f{}; m[i] = -1e30f; l[i] = 0.f; }

  const float scale = 0.07216878364870323f;  // 1/sqrt(192)
  unsigned short* sP = &sQ[w * 16 * 64];     // per-wave private 16x64

  for (int kb = 0; kb <= qb; ++kb) {
    __syncthreads();
    // ---- stage K (64 x 192) async gather ----
    for (int c = tid; c < 64 * 24; c += 128) {
      const int r = c / 24, d0 = (c % 24) * 8;
      const size_t tok = base + kb * 64 + r;
      const unsigned short* g = (d0 < 128)
          ? kvb + tok * 4096 + h * 256 + d0
          : krope + tok * 64 + (d0 - 128);
      async_copy16(&sK[r * 192 + d0], g);
    }
    // ---- stage V transposed [dv][key]: vector loads + b16 scatter ----
    for (int c = tid; c < 64 * 16; c += 128) {            // 16 x 8-elem per key
      const int key = c >> 4, d0 = (c & 15) * 8;
      const size_t tok = base + kb * 64 + key;
      union { u32x4 v; unsigned short s[8]; } t;
      t.v = *(const u32x4*)(kvb + tok * 4096 + h * 256 + 128 + d0);
#pragma unroll
      for (int j = 0; j < 8; ++j) sV[(d0 + j) * 64 + key] = t.s[j];
    }
    wait_async();
    __syncthreads();

    // ---- scores S = Q K^T (16 x 64 per wave) ----
    v8f s[4];
#pragma unroll
    for (int nt = 0; nt < 4; ++nt) {
      v8f accs = v8f{};
      const int kcol = nt * 16 + lane15;
#pragma unroll
      for (int ck = 0; ck < 6; ++ck) {
        Frag bf;
        const unsigned short* p = &sK[kcol * 192 + ck * 32 + lo];
        bf.u[0] = *(const u32x4*)(p);
        bf.u[1] = *(const u32x4*)(p + 16);
        accs = __builtin_amdgcn_wmma_f32_16x16x32_bf16(
            false, qf[ck].v, false, bf.v, (short)0, accs, false, false);
      }
#pragma unroll
      for (int j = 0; j < 8; ++j) accs[j] *= scale;
      s[nt] = accs;
    }

    // ---- causal mask on diagonal block ----
    if (kb == qb) {
#pragma unroll
      for (int nt = 0; nt < 4; ++nt) {
        const int key = nt * 16 + lane15;
#pragma unroll
        for (int j = 0; j < 8; ++j) {
          const int qrow = w * 16 + j + 8 * hi;
          if (key > qrow) s[nt][j] = -1e30f;
        }
      }
    }

    // ---- row max (16-lane xor butterfly) + exp ----
    float mb[8];
#pragma unroll
    for (int j = 0; j < 8; ++j) {
      float v = fmaxf(fmaxf(s[0][j], s[1][j]), fmaxf(s[2][j], s[3][j]));
#pragma unroll
      for (int msk = 1; msk < 16; msk <<= 1) v = fmaxf(v, __shfl_xor(v, msk, 32));
      mb[j] = fmaxf(m[j], v);
    }
#pragma unroll
    for (int nt = 0; nt < 4; ++nt)
#pragma unroll
      for (int j = 0; j < 8; ++j)
        s[nt][j] = __expf(s[nt][j] - mb[j]);

    // ---- P -> LDS (C-layout -> row-major 16x64), same-wave region ----
#pragma unroll
    for (int nt = 0; nt < 4; ++nt) {
      const int key = nt * 16 + lane15;
#pragma unroll
      for (int j = 0; j < 8; ++j)
        sP[(j + 8 * hi) * 64 + key] = f32_to_bf16(s[nt][j]);
    }
    asm volatile("s_wait_dscnt 0" ::: "memory");  // same-wave write->read

    // ---- P fragments (A-layout) ----
    Frag pf[2];
#pragma unroll
    for (int pk = 0; pk < 2; ++pk) {
      const unsigned short* p = &sP[lane15 * 64 + pk * 32 + lo];
      pf[pk].u[0] = *(const u32x4*)(p);
      pf[pk].u[1] = *(const u32x4*)(p + 16);
    }

    // ---- row sums on the matrix pipe: rs = P @ ones ----
    v8f rs = v8f{};
    rs = __builtin_amdgcn_wmma_f32_16x16x32_bf16(
        false, pf[0].v, false, ones.v, (short)0, rs, false, false);
    rs = __builtin_amdgcn_wmma_f32_16x16x32_bf16(
        false, pf[1].v, false, ones.v, (short)0, rs, false, false);

    // ---- m/l update + O rescale ----
#pragma unroll
    for (int j = 0; j < 8; ++j) {
      const float fac = __expf(m[j] - mb[j]);
      l[j] = l[j] * fac + rs[j];
      m[j] = mb[j];
#pragma unroll
      for (int ov = 0; ov < 8; ++ov) o[ov][j] *= fac;
    }

    // ---- O += P V (16 x 128) ----
#pragma unroll
    for (int ov = 0; ov < 8; ++ov) {
      const int vcol = ov * 16 + lane15;
#pragma unroll
      for (int pk = 0; pk < 2; ++pk) {
        Frag bf;
        const unsigned short* p = &sV[vcol * 64 + pk * 32 + lo];
        bf.u[0] = *(const u32x4*)(p);
        bf.u[1] = *(const u32x4*)(p + 16);
        o[ov] = __builtin_amdgcn_wmma_f32_16x16x32_bf16(
            false, pf[pk].v, false, bf.v, (short)0, o[ov], false, false);
      }
    }
  }

  // ---- normalize, store bf16 ----
#pragma unroll
  for (int ov = 0; ov < 8; ++ov) {
    const int col = h * 128 + ov * 16 + lane15;
#pragma unroll
    for (int j = 0; j < 8; ++j) {
      const size_t tok = base + qb * 64 + w * 16 + j + 8 * hi;
      out[tok * 2048 + col] = f32_to_bf16(o[ov][j] / l[j]);
    }
  }
}

// ---------------------------------------------------------------------------
// Host-side launcher
// ---------------------------------------------------------------------------
extern "C" void kernel_launch(void* const* d_in, const int* in_sizes, int n_in,
                              void* d_out, int out_size, void* d_ws, size_t ws_size,
                              hipStream_t stream)
{
  (void)in_sizes; (void)n_in; (void)out_size; (void)ws_size;
  const int B = 2, S = 2048, M = B * S;
  typedef unsigned short bf16_t;

  const float* hidden = (const float*)d_in[0];   // [M,2048] f32
  const float* wq_a   = (const float*)d_in[1];   // [2048,1536]
  const float* gq_a   = (const float*)d_in[2];   // [1536]
  const float* wq_b   = (const float*)d_in[3];   // [1536,3072]
  const float* wkv_a  = (const float*)d_in[4];   // [2048,576]
  const float* gkv_a  = (const float*)d_in[5];   // [512]
  const float* wkv_b  = (const float*)d_in[6];   // [512,4096]
  const float* wo     = (const float*)d_in[7];   // [2048,2048]

  char* ws = (char*)d_ws;
  bf16_t* q_a   = (bf16_t*)ws; ws += (size_t)M * 1536 * 2;
  bf16_t* kvbuf = (bf16_t*)ws; ws += (size_t)M * 576 * 2;
  bf16_t* c_kv  = (bf16_t*)ws; ws += (size_t)M * 512 * 2;
  bf16_t* kr    = (bf16_t*)ws; ws += (size_t)M * 64 * 2;
  bf16_t* qbuf  = (bf16_t*)ws; ws += (size_t)M * 3072 * 2;
  bf16_t* kvb   = (bf16_t*)ws; ws += (size_t)M * 4096 * 2;
  bf16_t* attn  = (bf16_t*)ws; ws += (size_t)M * 2048 * 2;

  const dim3 blk(256);
  // hidden @ wq_a -> q_a(bf16) ; hidden @ wkv_a -> kvbuf(bf16)
  gemm_wmma<float, true><<<dim3(1536 / 128, M / 128), blk, 0, stream>>>(
      hidden, wq_a, q_a, 1536, 2048, 2048, 1536, 1536);
  gemm_wmma<float, true><<<dim3((576 + 127) / 128, M / 128), blk, 0, stream>>>(
      hidden, wkv_a, kvbuf, 576, 2048, 2048, 576, 576);
  // rmsnorms (bf16 in/out, f32 math)
  rmsnorm_bf16<<<M, 256, 0, stream>>>(q_a, gq_a, q_a, 1536, 1536, 1536);
  rmsnorm_bf16<<<M, 256, 0, stream>>>(kvbuf, gkv_a, c_kv, 512, 576, 512);
  // k-rope from raw kv tail
  rope_k_kernel<<<M, 32, 0, stream>>>(kvbuf, kr, S);
  // q_lr @ wq_b -> qbuf(bf16) ; rope on q tail
  gemm_wmma<bf16_t, true><<<dim3(3072 / 128, M / 128), blk, 0, stream>>>(
      q_a, wq_b, qbuf, 3072, 1536, 1536, 3072, 3072);
  rope_q_kernel<<<M, 512, 0, stream>>>(qbuf, S);
  // c_kv @ wkv_b -> kvb(bf16)
  gemm_wmma<bf16_t, true><<<dim3(4096 / 128, M / 128), blk, 0, stream>>>(
      c_kv, wkv_b, kvb, 4096, 512, 512, 4096, 4096);
  // attention -> attn(bf16)
  mla_attn_kernel<<<dim3(S / 64, 16, B), dim3(128), 0, stream>>>(
      qbuf, kvb, kr, attn, S);
  // attn @ wo -> d_out (f32)
  gemm_wmma<bf16_t, false><<<dim3(2048 / 128, M / 128), blk, 0, stream>>>(
      attn, wo, (float*)d_out, 2048, 2048, 2048, 2048, 2048);
}
